// CombinedModel_3478923510081
// MI455X (gfx1250) — matrix operand order repaired and checked by reading, not compiled, not run on hardware
//
#include <hip/hip_runtime.h>

typedef __attribute__((ext_vector_type(16))) __bf16 v16bf;
typedef __attribute__((ext_vector_type(8)))  float  v8f;

#define NTOT 6912
#define ROWS_PER_G 432

__device__ inline v8f wmma_bf16(v16bf a, v16bf b, v8f c) {
  return __builtin_amdgcn_wmma_f32_16x16x32_bf16(false, a, false, b, (short)0, c, false, false);
}

// A fragment: 16x32 (MxK). lane&15 = M row; lanes>=16 hold K+8 groups.
template <typename T>
__device__ inline v16bf load_a(const T* X, int ld, int k0, int kmax, int lane) {
  int m = lane & 15;
  int kg = (lane >> 4) * 8;
  const T* row = X + m * ld;
  v16bf a;
#pragma unroll
  for (int j = 0; j < 8; ++j) {
    int kb = ((j < 4) ? (2 * j) : (16 + 2 * (j - 4))) + kg + k0;
    float f0 = (kb < kmax) ? (float)row[kb] : 0.0f;
    float f1 = (kb + 1 < kmax) ? (float)row[kb + 1] : 0.0f;
    a[2 * j] = (__bf16)f0;
    a[2 * j + 1] = (__bf16)f1;
  }
  return a;
}

// B fragment: 32x16 (KxN), row-major source W[k*ld + n]. lane&15 = N; element j = K (same K order as A).
__device__ inline v16bf load_b(const __bf16* W, int ld, int k0, int kmax, int lane) {
  int n = lane & 15;
  int kg = (lane >> 4) * 8;
  v16bf b;
#pragma unroll
  for (int j = 0; j < 16; ++j) {
    int k = k0 + kg + ((j < 8) ? j : (16 + (j - 8)));
    b[j] = (k < kmax) ? W[(size_t)k * ld + n] : (__bf16)0.0f;
  }
  return b;
}

// ---------------- weight f32 -> bf16 pre-conversion ----------------
__global__ void __launch_bounds__(256) cvt_kernel(const float* __restrict__ s,
                                                  __bf16* __restrict__ d, int n) {
  int i = (blockIdx.x * 256 + threadIdx.x) * 4;
  if (i + 3 < n) {
    float4 f = *(const float4*)(s + i);
    d[i] = (__bf16)f.x; d[i + 1] = (__bf16)f.y;
    d[i + 2] = (__bf16)f.z; d[i + 3] = (__bf16)f.w;
  }
}

// ---------------- SPP kernel ----------------
struct SppP {
  const __bf16* w1[3]; const __bf16* w2[3];
  const __bf16 *wm, *wf1, *wf2;
  const float* b1[3]; const float* b2[3];
  const float *bm, *bf1, *bf2, *ng, *nb, *ls;
};

__global__ void __launch_bounds__(256) spp_kernel(const float* __restrict__ h,
                                                  float* __restrict__ hs_f32,
                                                  __bf16* __restrict__ hs_bf16, SppP P) {
  __shared__ float  Xf[16 * 128];     // input rows, later h_mid
  __shared__ __bf16 Xbf[16 * 128];    // bf16 copy of input for stage-1 A fragments
  __shared__ __bf16 mids[16 * 112];   // bottleneck mids (16+32+64)
  __shared__ __bf16 outs[16 * 384];   // concat of three bottleneck outputs
  __shared__ __bf16 hn[16 * 128];
  __shared__ __bf16 y1[16 * 256];

  const int tid = threadIdx.x, lane = tid & 31, wave = tid >> 5;
  const int row0 = blockIdx.x * 16;
  const int n = lane & 15, mh = (lane >> 4) * 8;

  for (int i = tid; i < 16 * 128; i += 256) {
    float x = h[(size_t)row0 * 128 + i];
    Xf[i] = x;
    Xbf[i] = (__bf16)x;
  }
  __syncthreads();

  // stage 1: mid_i = relu(X @ W1_i + b1_i); 7 tiles; 2 interleaved K-chains
  if (wave < 7) {
    int bi = (wave == 0) ? 0 : (wave < 3 ? 1 : 2);
    int ct = (wave == 0) ? 0 : (wave < 3 ? wave - 1 : wave - 3);
    int bd = 16 << bi;
    int moff = (bi == 0) ? 0 : (bi == 1 ? 16 : 48);
    v8f ac0 = {}, ac1 = {};
#pragma unroll
    for (int k0 = 0; k0 < 128; k0 += 64) {
      ac0 = wmma_bf16(load_a(Xbf, 128, k0, 128, lane),
                      load_b(P.w1[bi] + ct * 16, bd, k0, 128, lane), ac0);
      ac1 = wmma_bf16(load_a(Xbf, 128, k0 + 32, 128, lane),
                      load_b(P.w1[bi] + ct * 16, bd, k0 + 32, 128, lane), ac1);
    }
    v8f acc = ac0 + ac1;
    float bias = P.b1[bi][ct * 16 + n];
#pragma unroll
    for (int j = 0; j < 8; ++j)
      mids[(mh + j) * 112 + moff + ct * 16 + n] = (__bf16)fmaxf(acc[j] + bias, 0.0f);
  }
  __syncthreads();

  // stage 2: out_i = relu(mid_i @ W2_i + b2_i); 24 tiles (short K chains)
  for (int job = wave; job < 24; job += 8) {
    int bi = job >> 3, ct = job & 7;
    int bd = 16 << bi;
    int moff = (bi == 0) ? 0 : (bi == 1 ? 16 : 48);
    v8f acc = {};
    for (int k0 = 0; k0 < bd; k0 += 32) {
      v16bf a = load_a(mids + moff, 112, k0, bd, lane);
      v16bf b = load_b(P.w2[bi] + ct * 16, 128, k0, bd, lane);
      acc = wmma_bf16(a, b, acc);
    }
    float bias = P.b2[bi][ct * 16 + n];
#pragma unroll
    for (int j = 0; j < 8; ++j)
      outs[(mh + j) * 384 + bi * 128 + ct * 16 + n] = (__bf16)fmaxf(acc[j] + bias, 0.0f);
  }
  __syncthreads();

  // stage 3: h_mid = X + ls * (outs @ Wm + bm); 8 tiles, K=384; 2 interleaved chains
  {
    int ct = wave;
    v8f ac0 = {}, ac1 = {};
    for (int k0 = 0; k0 < 384; k0 += 64) {
      ac0 = wmma_bf16(load_a(outs, 384, k0, 384, lane),
                      load_b(P.wm + ct * 16, 128, k0, 384, lane), ac0);
      ac1 = wmma_bf16(load_a(outs, 384, k0 + 32, 384, lane),
                      load_b(P.wm + ct * 16, 128, k0 + 32, 384, lane), ac1);
    }
    v8f acc = ac0 + ac1;
    float ls = P.ls[0];
    float bias = P.bm[ct * 16 + n];
#pragma unroll
    for (int j = 0; j < 8; ++j) {
      int idx = (mh + j) * 128 + ct * 16 + n;
      Xf[idx] = Xf[idx] + ls * (acc[j] + bias);
    }
  }
  __syncthreads();

  // stage 4: LayerNorm(h_mid) -> hn (bf16); 16 threads per row
  {
    int row = tid >> 4, sub = tid & 15;
    float s = 0.f, ss = 0.f;
#pragma unroll
    for (int i = 0; i < 8; ++i) {
      float x = Xf[row * 128 + sub + 16 * i];
      s += x; ss += x * x;
    }
#pragma unroll
    for (int m = 1; m < 16; m <<= 1) { s += __shfl_xor(s, m, 16); ss += __shfl_xor(ss, m, 16); }
    float mean = s / 128.0f;
    float inv = rsqrtf(ss / 128.0f - mean * mean + 1e-5f);
#pragma unroll
    for (int i = 0; i < 8; ++i) {
      int c = sub + 16 * i;
      hn[row * 128 + c] = (__bf16)((Xf[row * 128 + c] - mean) * inv * P.ng[c] + P.nb[c]);
    }
  }
  __syncthreads();

  // stage 5: y1 = relu(hn @ Wf1 + bf1); 16 tiles = 2 per wave, shared A, 2 accs
  {
    int t0 = wave, t1 = wave + 8;
    v8f ac0 = {}, ac1 = {};
    for (int k0 = 0; k0 < 128; k0 += 32) {
      v16bf a = load_a(hn, 128, k0, 128, lane);
      ac0 = wmma_bf16(a, load_b(P.wf1 + t0 * 16, 256, k0, 128, lane), ac0);
      ac1 = wmma_bf16(a, load_b(P.wf1 + t1 * 16, 256, k0, 128, lane), ac1);
    }
    float b0 = P.bf1[t0 * 16 + n], b1 = P.bf1[t1 * 16 + n];
#pragma unroll
    for (int j = 0; j < 8; ++j) {
      y1[(mh + j) * 256 + t0 * 16 + n] = (__bf16)fmaxf(ac0[j] + b0, 0.0f);
      y1[(mh + j) * 256 + t1 * 16 + n] = (__bf16)fmaxf(ac1[j] + b1, 0.0f);
    }
  }
  __syncthreads();

  // stage 6: out = h_mid + y1 @ Wf2 + bf2; 8 tiles, K=256; 2 interleaved chains
  {
    int ct = wave;
    v8f ac0 = {}, ac1 = {};
    for (int k0 = 0; k0 < 256; k0 += 64) {
      ac0 = wmma_bf16(load_a(y1, 256, k0, 256, lane),
                      load_b(P.wf2 + ct * 16, 128, k0, 256, lane), ac0);
      ac1 = wmma_bf16(load_a(y1, 256, k0 + 32, 256, lane),
                      load_b(P.wf2 + ct * 16, 128, k0 + 32, 256, lane), ac1);
    }
    v8f acc = ac0 + ac1;
    float bias = P.bf2[ct * 16 + n];
#pragma unroll
    for (int j = 0; j < 8; ++j) {
      int r = mh + j, c = ct * 16 + n;
      float o = Xf[r * 128 + c] + acc[j] + bias;
      hs_f32[(size_t)(row0 + r) * 128 + c] = o;
      hs_bf16[(size_t)(row0 + r) * 128 + c] = (__bf16)o;
    }
  }
}

// ---------------- QKV projections ----------------
// block = 16-row tile; wave = column tile; shared A feeds 3 interleaved chains (q,k,v).
struct QkvP { const __bf16 *wql, *wkp, *wvp, *wqp, *wkl, *wvl; };

__global__ void __launch_bounds__(256) qkv_kernel(const __bf16* __restrict__ hs,
                                                  __bf16* __restrict__ q,
                                                  __bf16* __restrict__ ktT,
                                                  __bf16* __restrict__ v, QkvP P) {
  int lane = threadIdx.x & 31, ct = threadIdx.x >> 5;
  int row0 = blockIdx.x * 16;
  bool is_lig = (row0 % ROWS_PER_G) >= 384;
  const __bf16* X = hs + (size_t)row0 * 128;
  const __bf16* Wq = (is_lig ? P.wql : P.wqp) + ct * 16;
  const __bf16* Wk = (is_lig ? P.wkl : P.wkp) + ct * 16;
  const __bf16* Wv = (is_lig ? P.wvl : P.wvp) + ct * 16;
  v8f aq = {}, ak = {}, av = {};
#pragma unroll
  for (int k0 = 0; k0 < 128; k0 += 32) {
    v16bf a = load_a(X, 128, k0, 128, lane);
    aq = wmma_bf16(a, load_b(Wq, 128, k0, 128, lane), aq);
    ak = wmma_bf16(a, load_b(Wk, 128, k0, 128, lane), ak);
    av = wmma_bf16(a, load_b(Wv, 128, k0, 128, lane), av);
  }
  int n = lane & 15, mh = (lane >> 4) * 8;
#pragma unroll
  for (int j = 0; j < 8; ++j) {
    // Q pre-scaled by 1/sqrt(AD)
    q[(size_t)(row0 + mh + j) * 128 + ct * 16 + n] = (__bf16)(aq[j] * 0.25f);
    // K transposed: ktT[dim * NTOT + row] (contiguous per lane)
    ktT[(size_t)(ct * 16 + n) * NTOT + row0 + mh + j] = (__bf16)ak[j];
    v[(size_t)(row0 + mh + j) * 128 + ct * 16 + n] = (__bf16)av[j];
  }
}

// ---------------- dense bipartite cross-attention ----------------
// block = (graph, head, 16-query-row tile). Scores + softmax + A@V in LDS.
__global__ void __launch_bounds__(256) attn_kernel(const __bf16* __restrict__ q,
                                                   const __bf16* __restrict__ ktT,
                                                   const __bf16* __restrict__ vv,
                                                   float* __restrict__ ctx,
                                                   int Mq, int Nkv, int qoff, int kvoff) {
  __shared__ float  S[16 * 384];
  __shared__ __bf16 A[16 * 384];
  __shared__ float  part[8][256];
  int mtiles = Mq >> 4;
  int bid = blockIdx.x;
  int mt = bid % mtiles;
  int gh = bid / mtiles;
  int head = gh & 7;
  int g = gh >> 3;
  int lane = threadIdx.x & 31, wave = threadIdx.x >> 5;
  int hc = head * 16;
  int n = lane & 15, mh = (lane >> 4) * 8;
  size_t qrow = (size_t)(g * ROWS_PER_G + qoff + mt * 16);
  size_t kvrow = (size_t)(g * ROWS_PER_G + kvoff);

  // scores: S = Qh @ Kh^T (K=16 padded to 32); up to 3 independent tiles per wave
  {
    int ntiles = Nkv >> 4;
    v16bf afrag = load_a(q + qrow * 128 + hc, 128, 0, 16, lane);
    int nt0 = wave, nt1 = wave + 8, nt2 = wave + 16;
    v8f s0 = {}, s1 = {}, s2 = {};
    const __bf16* kbase = ktT + (size_t)hc * NTOT + kvrow;
    if (nt0 < ntiles) s0 = wmma_bf16(afrag, load_b(kbase + nt0 * 16, NTOT, 0, 16, lane), s0);
    if (nt1 < ntiles) s1 = wmma_bf16(afrag, load_b(kbase + nt1 * 16, NTOT, 0, 16, lane), s1);
    if (nt2 < ntiles) s2 = wmma_bf16(afrag, load_b(kbase + nt2 * 16, NTOT, 0, 16, lane), s2);
#pragma unroll
    for (int j = 0; j < 8; ++j) {
      if (nt0 < ntiles) S[(mh + j) * Nkv + nt0 * 16 + n] = s0[j];
      if (nt1 < ntiles) S[(mh + j) * Nkv + nt1 * 16 + n] = s1[j];
      if (nt2 < ntiles) S[(mh + j) * Nkv + nt2 * 16 + n] = s2[j];
    }
  }
  __syncthreads();

  // row softmax (fp32), write bf16 attention weights
  for (int r = wave; r < 16; r += 8) {
    float mx = -1e30f;
    for (int c = lane; c < Nkv; c += 32) mx = fmaxf(mx, S[r * Nkv + c]);
#pragma unroll
    for (int m = 1; m < 32; m <<= 1) mx = fmaxf(mx, __shfl_xor(mx, m, 32));
    float sum = 0.f;
    for (int c = lane; c < Nkv; c += 32) {
      float e = __expf(S[r * Nkv + c] - mx);
      S[r * Nkv + c] = e;
      sum += e;
    }
#pragma unroll
    for (int m = 1; m < 32; m <<= 1) sum += __shfl_xor(sum, m, 32);
    float inv = 1.0f / sum;
    for (int c = lane; c < Nkv; c += 32) A[r * Nkv + c] = (__bf16)(S[r * Nkv + c] * inv);
  }
  __syncthreads();

  // ctx tile: A(16 x Nkv) @ Vh(Nkv x 16); K split across all 8 waves + LDS reduce
  {
    int nchunks = (Nkv + 31) / 32;
    v8f acc = {};
    for (int c = wave; c < nchunks; c += 8) {
      int k0 = c * 32;
      v16bf a = load_a(A, Nkv, k0, Nkv, lane);
      v16bf b = load_b(vv + kvrow * 128 + hc, 128, k0, Nkv, lane);
      acc = wmma_bf16(a, b, acc);
    }
#pragma unroll
    for (int j = 0; j < 8; ++j) part[wave][(mh + j) * 16 + n] = acc[j];
  }
  __syncthreads();
  {
    int t = threadIdx.x;             // m = t>>4, n = t&15
    float s = 0.f;
#pragma unroll
    for (int w = 0; w < 8; ++w) s += part[w][t];
    ctx[(size_t)(g * Mq + mt * 16 + (t >> 4)) * 128 + hc + (t & 15)] = s;
  }
}

// ---------------- gated update + output LN/FFN ----------------
struct OutP {
  const __bf16 *wg, *wu, *wf1, *wf2;
  const float *bg, *bu, *ng, *nb, *bf1, *bf2;
};

__global__ void __launch_bounds__(256) update_kernel(const float* __restrict__ hs,
                                                     const float* __restrict__ ctx,
                                                     float* __restrict__ out, OutP P,
                                                     int Mside, int roff) {
  __shared__ __bf16 xcat[16 * 256];   // [x, ctx] bf16
  __shared__ float  Xf[16 * 128];     // x, later xu
  __shared__ float  G[16 * 128];
  __shared__ float  U[16 * 128];
  __shared__ __bf16 hn[16 * 128];
  __shared__ __bf16 y1[16 * 512];

  int tid = threadIdx.x, lane = tid & 31, wave = tid >> 5;
  int n = lane & 15, mh = (lane >> 4) * 8;
  int crow0 = blockIdx.x * 16;
  int g = crow0 / Mside;
  int local = crow0 % Mside;
  size_t grow0 = (size_t)g * ROWS_PER_G + roff + local;

  for (int i = tid; i < 16 * 128; i += 256) {
    int r = i >> 7, c = i & 127;
    float x = hs[(grow0 + r) * 128 + c];
    Xf[i] = x;
    xcat[r * 256 + c] = (__bf16)x;
    xcat[r * 256 + 128 + c] = (__bf16)ctx[(size_t)(crow0 + r) * 128 + c];
  }
  __syncthreads();

  // gate (K=256) and update (K=128): 3 interleaved chains per wave;
  // gate's upper-half A fragments (ctx part of xcat) are exactly upd's A.
  {
    int ct = wave;
    v8f accg0 = {}, accg1 = {}, accu = {};
#pragma unroll
    for (int i = 0; i < 4; ++i) {
      v16bf alo = load_a(xcat, 256, i * 32, 256, lane);
      v16bf ahi = load_a(xcat, 256, 128 + i * 32, 256, lane);
      accg0 = wmma_bf16(alo, load_b(P.wg + ct * 16, 128, i * 32, 256, lane), accg0);
      accg1 = wmma_bf16(ahi, load_b(P.wg + ct * 16, 128, 128 + i * 32, 256, lane), accg1);
      accu  = wmma_bf16(ahi, load_b(P.wu + ct * 16, 128, i * 32, 128, lane), accu);
    }
    v8f accg = accg0 + accg1;
    float bg = P.bg[ct * 16 + n], bu = P.bu[ct * 16 + n];
#pragma unroll
    for (int j = 0; j < 8; ++j) {
      G[(mh + j) * 128 + ct * 16 + n] = 1.0f / (1.0f + __expf(-(accg[j] + bg)));
      U[(mh + j) * 128 + ct * 16 + n] = accu[j] + bu;
    }
  }
  __syncthreads();

  for (int i = tid; i < 16 * 128; i += 256) Xf[i] = Xf[i] + G[i] * U[i];
  __syncthreads();

  // LayerNorm
  {
    int row = tid >> 4, sub = tid & 15;
    float s = 0.f, ss = 0.f;
#pragma unroll
    for (int i = 0; i < 8; ++i) {
      float x = Xf[row * 128 + sub + 16 * i];
      s += x; ss += x * x;
    }
#pragma unroll
    for (int m = 1; m < 16; m <<= 1) { s += __shfl_xor(s, m, 16); ss += __shfl_xor(ss, m, 16); }
    float mean = s / 128.0f;
    float inv = rsqrtf(ss / 128.0f - mean * mean + 1e-5f);
#pragma unroll
    for (int i = 0; i < 8; ++i) {
      int c = sub + 16 * i;
      hn[row * 128 + c] = (__bf16)((Xf[row * 128 + c] - mean) * inv * P.ng[c] + P.nb[c]);
    }
  }
  __syncthreads();

  // ffn1: 128 -> 512 relu; 32 tiles = 2 pair-jobs per wave, shared A, 2 accs
  for (int jp = wave; jp < 16; jp += 8) {
    int t0 = jp, t1 = jp + 16;
    v8f ac0 = {}, ac1 = {};
    for (int k0 = 0; k0 < 128; k0 += 32) {
      v16bf a = load_a(hn, 128, k0, 128, lane);
      ac0 = wmma_bf16(a, load_b(P.wf1 + t0 * 16, 512, k0, 128, lane), ac0);
      ac1 = wmma_bf16(a, load_b(P.wf1 + t1 * 16, 512, k0, 128, lane), ac1);
    }
    float b0 = P.bf1[t0 * 16 + n], b1 = P.bf1[t1 * 16 + n];
#pragma unroll
    for (int j = 0; j < 8; ++j) {
      y1[(mh + j) * 512 + t0 * 16 + n] = (__bf16)fmaxf(ac0[j] + b0, 0.0f);
      y1[(mh + j) * 512 + t1 * 16 + n] = (__bf16)fmaxf(ac1[j] + b1, 0.0f);
    }
  }
  __syncthreads();

  // ffn2: 512 -> 128, residual; K=512 split into 4 interleaved chains
  {
    int ct = wave;
    v8f a0 = {}, a1 = {}, a2 = {}, a3 = {};
    for (int k0 = 0; k0 < 512; k0 += 128) {
      a0 = wmma_bf16(load_a(y1, 512, k0, 512, lane),
                     load_b(P.wf2 + ct * 16, 128, k0, 512, lane), a0);
      a1 = wmma_bf16(load_a(y1, 512, k0 + 32, 512, lane),
                     load_b(P.wf2 + ct * 16, 128, k0 + 32, 512, lane), a1);
      a2 = wmma_bf16(load_a(y1, 512, k0 + 64, 512, lane),
                     load_b(P.wf2 + ct * 16, 128, k0 + 64, 512, lane), a2);
      a3 = wmma_bf16(load_a(y1, 512, k0 + 96, 512, lane),
                     load_b(P.wf2 + ct * 16, 128, k0 + 96, 512, lane), a3);
    }
    v8f acc = (a0 + a1) + (a2 + a3);
    float bias = P.bf2[ct * 16 + n];
#pragma unroll
    for (int j = 0; j < 8; ++j) {
      int r = mh + j, c = ct * 16 + n;
      out[(grow0 + r) * 128 + c] = Xf[r * 128 + c] + acc[j] + bias;
    }
  }
}

extern "C" void kernel_launch(void* const* d_in, const int* in_sizes, int n_in,
                              void* d_out, int out_size, void* d_ws, size_t ws_size,
                              hipStream_t stream) {
  const float* h = (const float*)d_in[0];
  auto F = [&](int i) { return (const float*)d_in[i]; };

  char* w = (char*)d_ws;
  float*  hs_f32 = (float*)w;   w += (size_t)NTOT * 128 * sizeof(float);
  __bf16* hs_bf  = (__bf16*)w;  w += (size_t)NTOT * 128 * sizeof(__bf16);
  __bf16* qb     = (__bf16*)w;  w += (size_t)NTOT * 128 * sizeof(__bf16);
  __bf16* ktT    = (__bf16*)w;  w += (size_t)NTOT * 128 * sizeof(__bf16);
  __bf16* vb     = (__bf16*)w;  w += (size_t)NTOT * 128 * sizeof(__bf16);
  float*  ctx_l  = (float*)w;   w += (size_t)768 * 128 * sizeof(float);
  float*  ctx_p  = (float*)w;   w += (size_t)6144 * 128 * sizeof(float);
  __bf16* cur    = (__bf16*)w;  // bf16 weight arena (~1.2 MB)

  auto CVT = [&](int srcIdx, int n) -> const __bf16* {
    __bf16* dst = cur;
    cur += n;
    cvt_kernel<<<(n + 1023) / 1024, 256, 0, stream>>>((const float*)d_in[srcIdx], dst, n);
    return dst;
  };

  SppP sp;
  sp.w1[0] = CVT(7, 128 * 16);   sp.b1[0] = F(8);
  sp.w2[0] = CVT(9, 16 * 128);   sp.b2[0] = F(10);
  sp.w1[1] = CVT(11, 128 * 32);  sp.b1[1] = F(12);
  sp.w2[1] = CVT(13, 32 * 128);  sp.b2[1] = F(14);
  sp.w1[2] = CVT(15, 128 * 64);  sp.b1[2] = F(16);
  sp.w2[2] = CVT(17, 64 * 128);  sp.b2[2] = F(18);
  sp.wm  = CVT(19, 384 * 128);   sp.bm  = F(20);
  sp.wf1 = CVT(21, 128 * 256);   sp.bf1 = F(22);
  sp.wf2 = CVT(23, 256 * 128);   sp.bf2 = F(24);
  sp.ng = F(25); sp.nb = F(26); sp.ls = F(27);

  QkvP qp;
  qp.wql = CVT(28, 128 * 128); qp.wkp = CVT(29, 128 * 128); qp.wvp = CVT(30, 128 * 128);
  qp.wqp = CVT(31, 128 * 128); qp.wkl = CVT(32, 128 * 128); qp.wvl = CVT(33, 128 * 128);

  OutP ol;
  ol.wg  = CVT(34, 256 * 128); ol.bg  = F(35);
  ol.wu  = CVT(36, 128 * 128); ol.bu  = F(37);
  ol.ng = F(38); ol.nb = F(39);
  ol.wf1 = CVT(40, 128 * 512); ol.bf1 = F(41);
  ol.wf2 = CVT(42, 512 * 128); ol.bf2 = F(43);

  OutP op;
  op.wg  = CVT(44, 256 * 128); op.bg  = F(45);
  op.wu  = CVT(46, 128 * 128); op.bu  = F(47);
  op.ng = F(48); op.nb = F(49);
  op.wf1 = CVT(50, 128 * 512); op.bf1 = F(51);
  op.wf2 = CVT(52, 512 * 128); op.bf2 = F(53);

  spp_kernel<<<432, 256, 0, stream>>>(h, hs_f32, hs_bf, sp);
  qkv_kernel<<<432, 256, 0, stream>>>(hs_bf, qb, ktT, vb, qp);
  // ligand queries over protein keys/values: Mq=48, Nkv=384
  attn_kernel<<<16 * 8 * 3, 256, 0, stream>>>(qb, ktT, vb, ctx_l, 48, 384, 384, 0);
  // protein queries over ligand keys/values: Mq=384, Nkv=48
  attn_kernel<<<16 * 8 * 24, 256, 0, stream>>>(qb, ktT, vb, ctx_p, 384, 48, 0, 384);
  update_kernel<<<48, 256, 0, stream>>>(hs_f32, ctx_l, (float*)d_out, ol, 48, 384);
  update_kernel<<<384, 256, 0, stream>>>(hs_f32, ctx_p, (float*)d_out, op, 384, 0);
}
